// TransformerBlock_1821066133604
// MI455X (gfx1250) — matrix operand order repaired and checked by reading, not compiled, not run on hardware
//
#include <hip/hip_runtime.h>
#include <cstdint>
#include <cstddef>

#define EMBED 768
#define NHEAD 12
#define HDIM  64
#define DFC   3072
#define BATCH 2
#define SEQ   2048
#define MROWS (BATCH*SEQ)   /* 4096 */

typedef __bf16 bf16;
typedef __attribute__((ext_vector_type(16))) __bf16 v16bf;
typedef __attribute__((ext_vector_type(8)))  __bf16 v8bf;
typedef __attribute__((ext_vector_type(8)))  float  v8f;
typedef __attribute__((ext_vector_type(4)))  int    v4i_t;

// ---- gfx1250 async memory->LDS path (guarded; falls back to sync staging) ----
#if defined(__AMDGCN__) && __has_builtin(__builtin_amdgcn_global_load_async_to_lds_b128) && __has_builtin(__builtin_amdgcn_s_wait_asynccnt)
#define HAVE_ASYNC 1
typedef __attribute__((address_space(1))) v4i_t as1_v4i;
typedef __attribute__((address_space(3))) v4i_t as3_v4i;
#define ASYNC_B128(g, l) __builtin_amdgcn_global_load_async_to_lds_b128((as1_v4i*)(v4i_t*)(g), (as3_v4i*)(v4i_t*)(l), 0, 0)
#define ASYNC_WAIT(n)    __builtin_amdgcn_s_wait_asynccnt(n)
#else
#define HAVE_ASYNC 0
#define ASYNC_B128(g, l) (void)0
#define ASYNC_WAIT(n)    (void)0
#endif

__device__ __forceinline__ v16bf cat16(v8bf lo, v8bf hi) {
  v16bf r;
#pragma unroll
  for (int j = 0; j < 8; ++j) { r[j] = lo[j]; r[j + 8] = hi[j]; }
  return r;
}

// ---------------- casts ----------------
__global__ __launch_bounds__(256) void cast_bf16_kernel(const float* __restrict__ s,
                                                        bf16* __restrict__ d, int n) {
  int i = blockIdx.x * 256 + threadIdx.x;
  if (i < n) d[i] = (bf16)s[i];
}

// src [K,N] f32 row-major  ->  dst [N,K] bf16 row-major
__global__ __launch_bounds__(256) void transpose_cast_kernel(const float* __restrict__ s,
                                                             bf16* __restrict__ d,
                                                             int K, int N) {
  int i = blockIdx.x * 256 + threadIdx.x;
  if (i < K * N) {
    int k = i / N, n = i - k * N;
    d[(size_t)n * K + k] = (bf16)s[i];
  }
}

// ---------------- GEMM:  C[M,N] = A[M,K] x Bw[N,K]^T  (bf16 in, f32 accum) ----------
// Block 64x128, BK=64 (two WMMA k-substeps), 8 waves in 2x4, wave tile 32x32.
// Double-buffered LDS staged with GLOBAL_LOAD_ASYNC_TO_LDS_B128 when available.
// EPI: 0=f32 store, 1=bf16 store, 2=bf16 store transposed to [B,H,D,S] (for V),
//      3=bias+relu->bf16, 4=bias->f32
template<int EPI>
__global__ __launch_bounds__(256) void gemm_bf16_kernel(
    const bf16* __restrict__ A, const bf16* __restrict__ Bw,
    const float* __restrict__ bias,
    float* __restrict__ Cf, bf16* __restrict__ Cb,
    int M, int N, int K)
{
  __shared__ __align__(16) bf16 As[2][64][72];    // row stride 144B (16B multiple)
  __shared__ __align__(16) bf16 Bs[2][128][72];

  const int tid  = threadIdx.x;
  const int lane = tid & 31;
  const int wv   = tid >> 5;
  const int ln   = lane & 15;
  const int hi   = lane >> 4;
  const int wr   = wv >> 2;        // 0..1  (wave row in 2x4 grid)
  const int wc   = wv & 3;         // 0..3
  const int m_blk = blockIdx.y * 64;
  const int n_blk = blockIdx.x * 128;

  // cooperative staging indices (BK = 64)
  const int arow = tid >> 2;          // 0..63
  const int acol = (tid & 3) * 16;    // 0,16,32,48   (16 bf16 = 2 x b128 per thread)
  const int brow = tid >> 1;          // 0..127
  const int bcol = (tid & 1) * 32;    // 0,32         (32 bf16 = 4 x b128 per thread)

  v8f acc[2][2];
#pragma unroll
  for (int i = 0; i < 2; ++i)
#pragma unroll
    for (int j = 0; j < 2; ++j) acc[i][j] = (v8f){0,0,0,0,0,0,0,0};

  const bf16* aptr = A  + (size_t)(m_blk + arow) * K + acol;
  const bf16* bptr = Bw + (size_t)(n_blk + brow) * K + bcol;

#if HAVE_ASYNC
#define STAGE(buf, k0)                                                        \
  do {                                                                        \
    ASYNC_B128(aptr + (k0),      &As[buf][arow][acol]);                       \
    ASYNC_B128(aptr + (k0) + 8,  &As[buf][arow][acol + 8]);                   \
    ASYNC_B128(bptr + (k0),      &Bs[buf][brow][bcol]);                       \
    ASYNC_B128(bptr + (k0) + 8,  &Bs[buf][brow][bcol + 8]);                   \
    ASYNC_B128(bptr + (k0) + 16, &Bs[buf][brow][bcol + 16]);                  \
    ASYNC_B128(bptr + (k0) + 24, &Bs[buf][brow][bcol + 24]);                  \
  } while (0)
#else
#define STAGE(buf, k0)                                                        \
  do {                                                                        \
    *(v8bf*)&As[buf][arow][acol]      = *(const v8bf*)(aptr + (k0));          \
    *(v8bf*)&As[buf][arow][acol + 8]  = *(const v8bf*)(aptr + (k0) + 8);      \
    *(v8bf*)&Bs[buf][brow][bcol]      = *(const v8bf*)(bptr + (k0));          \
    *(v8bf*)&Bs[buf][brow][bcol + 8]  = *(const v8bf*)(bptr + (k0) + 8);      \
    *(v8bf*)&Bs[buf][brow][bcol + 16] = *(const v8bf*)(bptr + (k0) + 16);     \
    *(v8bf*)&Bs[buf][brow][bcol + 24] = *(const v8bf*)(bptr + (k0) + 24);     \
  } while (0)
#endif

  const int nk = K >> 6;              // K multiple of 64 for all our shapes
  STAGE(0, 0);
  for (int kt = 0; kt < nk; ++kt) {
    const int cur = kt & 1;
    if (kt + 1 < nk) {
      STAGE(cur ^ 1, (kt + 1) << 6);
      ASYNC_WAIT(6);                   // previous stage's 6 loads complete
    } else {
      ASYNC_WAIT(0);                   // final stage fully landed
    }
    __syncthreads();

#pragma unroll
    for (int ks = 0; ks < 2; ++ks) {     // two k=32 substeps within the stage
      v16bf af[2], bfg[2];
#pragma unroll
      for (int i = 0; i < 2; ++i) {
        // A fragment: lane l holds M = l&15, K = hi*16 + 0..15 (contiguous)
        const bf16* p = &As[cur][wr * 32 + i * 16 + ln][ks * 32];
        af[i] = cat16(*(const v8bf*)(p + hi * 16), *(const v8bf*)(p + hi * 16 + 8));
      }
#pragma unroll
      for (int j = 0; j < 2; ++j) {
        // B fragment: lane l holds N = l&15, K = hi*16 + 0..15 (contiguous)
        const bf16* p = &Bs[cur][wc * 32 + j * 16 + ln][ks * 32 + hi * 16];
        bfg[j] = cat16(*(const v8bf*)p, *(const v8bf*)(p + 8));
      }
#pragma unroll
      for (int i = 0; i < 2; ++i)
#pragma unroll
        for (int j = 0; j < 2; ++j)
          acc[i][j] = __builtin_amdgcn_wmma_f32_16x16x32_bf16(
              false, af[i], false, bfg[j], (short)0, acc[i][j], false, false);
    }
    __syncthreads();                     // protect buffer before next overwrite
  }
#undef STAGE

  // epilogue: C fragment layout -> row = r + 8*hi, col = l&15
#pragma unroll
  for (int i = 0; i < 2; ++i) {
#pragma unroll
    for (int j = 0; j < 2; ++j) {
      const int col = n_blk + wc * 32 + j * 16 + ln;
#pragma unroll
      for (int r = 0; r < 8; ++r) {
        const int row = m_blk + wr * 32 + i * 16 + hi * 8 + r;
        float v = acc[i][j][r];
        if (EPI == 3 || EPI == 4) v += bias[col];
        if (EPI == 3) v = fmaxf(v, 0.0f);
        if (EPI == 0 || EPI == 4) {
          Cf[(size_t)row * N + col] = v;
        } else if (EPI == 1 || EPI == 3) {
          Cb[(size_t)row * N + col] = (bf16)v;
        } else {
          // V stored transposed: [B][H][D][S] so P*V B-fragments are contiguous
          const int hh = col >> 6, dd = col & 63;
          const int bb = row >> 11, ss = row & (SEQ - 1);
          Cb[(((size_t)(bb * NHEAD + hh) << 6) + dd) * SEQ + ss] = (bf16)v;
        }
      }
    }
  }
}

// ---------------- causal flash attention (bf16 WMMA, fp32 softmax) ----------------
// Q,K: [B,S,E] bf16 ; Vt: [B,H,D,S] bf16 ; O: [B,S,E] bf16
// One wave handles one 16-query tile of one (b,h); 32 keys per iteration.
__global__ __launch_bounds__(256) void attention_kernel(
    const bf16* __restrict__ Q, const bf16* __restrict__ Km,
    const bf16* __restrict__ Vt, bf16* __restrict__ O)
{
  __shared__ __align__(16) bf16 pl[8][16][32];   // per-wave P staging (C->A relayout)

  const int lane = threadIdx.x & 31;
  const int wv   = threadIdx.x >> 5;
  const int ln   = lane & 15;
  const int hi   = lane >> 4;
  const int q0   = (blockIdx.x * 8 + wv) * 16;
  const int h    = blockIdx.y;
  const int b    = blockIdx.z;

  // Q A-fragments for the two K-dim chunks (d 0..31, 32..63)
  const bf16* qrow = Q + ((size_t)(b * SEQ) + q0 + ln) * EMBED + h * HDIM;
  v16bf aq[2];
#pragma unroll
  for (int c = 0; c < 2; ++c)
    aq[c] = cat16(*(const v8bf*)(qrow + c * 32 + hi * 16),
                  *(const v8bf*)(qrow + c * 32 + hi * 16 + 8));

  const bf16* kbase = Km + (size_t)(b * SEQ) * EMBED + h * HDIM;
  const bf16* vbase = Vt + ((size_t)(b * NHEAD + h) * HDIM) * SEQ;

  float mrow[8], lrow[8];
  v8f o[4];
#pragma unroll
  for (int r = 0; r < 8; ++r) { mrow[r] = -__builtin_inff(); lrow[r] = 0.0f; }
#pragma unroll
  for (int nc = 0; nc < 4; ++nc) o[nc] = (v8f){0,0,0,0,0,0,0,0};

  const int ktmax = (q0 + 15) >> 5;              // causal pruning
  for (int kt = 0; kt <= ktmax; ++kt) {
    const int k0 = kt * 32;
    v8f s0 = (v8f){0,0,0,0,0,0,0,0};
    v8f s1 = s0;
#pragma unroll
    for (int c = 0; c < 2; ++c) {
      // B-fragment of K^T: lane n = key (l&15), k = d = c*32 + hi*16 + 0..15
      const bf16* kp0 = kbase + (size_t)(k0 + ln) * EMBED + c * 32 + hi * 16;
      const bf16* kp1 = kbase + (size_t)(k0 + 16 + ln) * EMBED + c * 32 + hi * 16;
      v16bf bk0 = cat16(*(const v8bf*)kp0, *(const v8bf*)(kp0 + 8));
      v16bf bk1 = cat16(*(const v8bf*)kp1, *(const v8bf*)(kp1 + 8));
      s0 = __builtin_amdgcn_wmma_f32_16x16x32_bf16(false, aq[c], false, bk0, (short)0, s0, false, false);
      s1 = __builtin_amdgcn_wmma_f32_16x16x32_bf16(false, aq[c], false, bk1, (short)0, s1, false, false);
    }

    // scale 1/sqrt(64), causal mask, online softmax (row = q0 + hi*8 + r)
#pragma unroll
    for (int r = 0; r < 8; ++r) {
      const int row = q0 + hi * 8 + r;
      float v0 = s0[r] * 0.125f;
      float v1 = s1[r] * 0.125f;
      if (k0 + ln > row)      v0 = -__builtin_inff();
      if (k0 + 16 + ln > row) v1 = -__builtin_inff();
      float t = fmaxf(v0, v1);
#pragma unroll
      for (int msk = 1; msk < 16; msk <<= 1) t = fmaxf(t, __shfl_xor(t, msk, 32));
      const float mnew = fmaxf(mrow[r], t);
      const float corr = __expf(mrow[r] - mnew);
      const float p0 = __expf(v0 - mnew);
      const float p1 = __expf(v1 - mnew);
      float rs = p0 + p1;
#pragma unroll
      for (int msk = 1; msk < 16; msk <<= 1) rs += __shfl_xor(rs, msk, 32);
      lrow[r] = lrow[r] * corr + rs;
      mrow[r] = mnew;
#pragma unroll
      for (int nc = 0; nc < 4; ++nc) o[nc][r] *= corr;
      pl[wv][hi * 8 + r][ln]      = (bf16)p0;
      pl[wv][hi * 8 + r][16 + ln] = (bf16)p1;
    }

    // reload P as an A-fragment (16x32); same-wave LDS RAW ordered by DScnt
    const bf16* pp = &pl[wv][ln][0];
    v16bf ap = cat16(*(const v8bf*)(pp + hi * 16), *(const v8bf*)(pp + hi * 16 + 8));

    // O += P * V ; V B-frag contiguous because V is stored [B,H,D,S]
#pragma unroll
    for (int nc = 0; nc < 4; ++nc) {
      const bf16* vp = vbase + (size_t)(nc * 16 + ln) * SEQ + k0 + hi * 16;
      v16bf bv = cat16(*(const v8bf*)vp, *(const v8bf*)(vp + 8));
      o[nc] = __builtin_amdgcn_wmma_f32_16x16x32_bf16(false, ap, false, bv, (short)0, o[nc], false, false);
    }
  }

  // normalize and store
#pragma unroll
  for (int r = 0; r < 8; ++r) {
    const float inv = 1.0f / lrow[r];
    const int row = q0 + hi * 8 + r;
#pragma unroll
    for (int nc = 0; nc < 4; ++nc)
      O[((size_t)(b * SEQ) + row) * EMBED + h * HDIM + nc * 16 + ln] = (bf16)(o[nc][r] * inv);
  }
}

// ---------------- fused residual add + LayerNorm (ddof=1, eps=1e-5) ----------------
__global__ __launch_bounds__(256) void add_ln_kernel(
    const float* __restrict__ a, const float* __restrict__ res,
    const float* __restrict__ w, const float* __restrict__ bias,
    float* __restrict__ outf, bf16* __restrict__ outb)
{
  const int row = blockIdx.x;
  const int tid = threadIdx.x;
  const size_t base = (size_t)row * EMBED;
  float v[3];
  float sum = 0.f, ss = 0.f;
#pragma unroll
  for (int j = 0; j < 3; ++j) {
    const int i = tid + j * 256;
    v[j] = a[base + i] + res[base + i];
    sum += v[j]; ss += v[j] * v[j];
  }
#pragma unroll
  for (int msk = 1; msk < 32; msk <<= 1) {
    sum += __shfl_xor(sum, msk, 32);
    ss  += __shfl_xor(ss,  msk, 32);
  }
  __shared__ float s1[8], s2[8];
  if ((tid & 31) == 0) { s1[tid >> 5] = sum; s2[tid >> 5] = ss; }
  __syncthreads();
  float tot = 0.f, tots = 0.f;
#pragma unroll
  for (int j = 0; j < 8; ++j) { tot += s1[j]; tots += s2[j]; }
  const float mean = tot * (1.0f / EMBED);
  const float var  = (tots - (float)EMBED * mean * mean) * (1.0f / (EMBED - 1));
  const float rstd = rsqrtf(var + 1e-5f);
#pragma unroll
  for (int j = 0; j < 3; ++j) {
    const int i = tid + j * 256;
    const float y = (v[j] - mean) * rstd * w[i] + bias[i];
    outf[base + i] = y;
    if (outb) outb[base + i] = (bf16)y;
  }
}

// ---------------- launch ----------------
extern "C" void kernel_launch(void* const* d_in, const int* in_sizes, int n_in,
                              void* d_out, int out_size, void* d_ws, size_t ws_size,
                              hipStream_t stream) {
  (void)in_sizes; (void)n_in; (void)out_size; (void)ws_size;
  const float* x     = (const float*)d_in[0];
  const float* in_w  = (const float*)d_in[1];   // [3,E,E]
  const float* out_w = (const float*)d_in[2];   // [E,E]
  const float* ln1w  = (const float*)d_in[3];
  const float* ln1b  = (const float*)d_in[4];
  const float* fc1w  = (const float*)d_in[5];   // [F,E]
  const float* fc1b  = (const float*)d_in[6];
  const float* fc2w  = (const float*)d_in[7];   // [E,F]
  const float* fc2b  = (const float*)d_in[8];
  const float* ln2w  = (const float*)d_in[9];
  const float* ln2b  = (const float*)d_in[10];
  float* out = (float*)d_out;

  char* ws = (char*)d_ws;
  size_t off = 0;
  auto alloc = [&](size_t bytes) {
    char* p = ws + off;
    off = (off + bytes + 255) & ~(size_t)255;
    return p;
  };

  const size_t ME = (size_t)MROWS * EMBED;
  const size_t MF = (size_t)MROWS * DFC;
  const size_t EE = (size_t)EMBED * EMBED;
  const size_t FE = (size_t)DFC * EMBED;

  bf16*  xb   = (bf16*)alloc(ME * 2);
  bf16*  wqt  = (bf16*)alloc(EE * 2);
  bf16*  wkt  = (bf16*)alloc(EE * 2);
  bf16*  wvt  = (bf16*)alloc(EE * 2);
  bf16*  wot  = (bf16*)alloc(EE * 2);
  bf16*  f1b  = (bf16*)alloc(FE * 2);
  bf16*  f2b  = (bf16*)alloc(FE * 2);
  bf16*  qb   = (bf16*)alloc(ME * 2);
  bf16*  kb   = (bf16*)alloc(ME * 2);
  bf16*  vtb  = (bf16*)alloc(ME * 2);
  bf16*  attb = (bf16*)alloc(ME * 2);
  float* proj = (float*)alloc(ME * 4);
  float* mh   = (float*)alloc(ME * 4);
  bf16*  mhb  = (bf16*)alloc(ME * 2);
  bf16*  hb   = (bf16*)alloc(MF * 2);
  float* mlp  = proj;   // proj is dead after LN1; alias for MLP output

  dim3 blk(256);

  // casts / weight pre-transposes
  cast_bf16_kernel<<<(int)(ME / 256), blk, 0, stream>>>(x, xb, (int)ME);
  transpose_cast_kernel<<<(int)(EE / 256), blk, 0, stream>>>(in_w + 0 * EE, wqt, EMBED, EMBED);
  transpose_cast_kernel<<<(int)(EE / 256), blk, 0, stream>>>(in_w + 1 * EE, wkt, EMBED, EMBED);
  transpose_cast_kernel<<<(int)(EE / 256), blk, 0, stream>>>(in_w + 2 * EE, wvt, EMBED, EMBED);
  transpose_cast_kernel<<<(int)(EE / 256), blk, 0, stream>>>(out_w, wot, EMBED, EMBED);
  cast_bf16_kernel<<<(int)(FE / 256), blk, 0, stream>>>(fc1w, f1b, (int)FE);
  cast_bf16_kernel<<<(int)(FE / 256), blk, 0, stream>>>(fc2w, f2b, (int)FE);

  dim3 gE(EMBED / 128, MROWS / 64);   // (6, 64)
  dim3 gF(DFC / 128, MROWS / 64);     // (24, 64)

  // QKV projections (V stored transposed [B,H,D,S])
  gemm_bf16_kernel<1><<<gE, blk, 0, stream>>>(xb, wqt, nullptr, nullptr, qb,  MROWS, EMBED, EMBED);
  gemm_bf16_kernel<1><<<gE, blk, 0, stream>>>(xb, wkt, nullptr, nullptr, kb,  MROWS, EMBED, EMBED);
  gemm_bf16_kernel<2><<<gE, blk, 0, stream>>>(xb, wvt, nullptr, nullptr, vtb, MROWS, EMBED, EMBED);

  // causal attention
  attention_kernel<<<dim3(SEQ / 128, NHEAD, BATCH), blk, 0, stream>>>(qb, kb, vtb, attb);

  // output projection (f32), then residual + LN1 -> mh (f32) + mhb (bf16)
  gemm_bf16_kernel<0><<<gE, blk, 0, stream>>>(attb, wot, nullptr, proj, nullptr, MROWS, EMBED, EMBED);
  add_ln_kernel<<<MROWS, blk, 0, stream>>>(proj, x, ln1w, ln1b, mh, mhb);

  // MLP: fc1 (bias+relu -> bf16), fc2 (bias -> f32)
  gemm_bf16_kernel<3><<<gF, blk, 0, stream>>>(mhb, f1b, fc1b, nullptr, hb, MROWS, DFC,  EMBED);
  gemm_bf16_kernel<4><<<gE, blk, 0, stream>>>(hb,  f2b, fc2b, mlp, nullptr, MROWS, EMBED, DFC);

  // residual + LN2 -> final output (f32)
  add_ln_kernel<<<MROWS, blk, 0, stream>>>(mlp, mh, ln2w, ln2b, out, nullptr);
}
